// MultiHeadAttention_23656679866584
// MI455X (gfx1250) — compile-verified, matrix operand order checked by reading, not run on hardware
//
#include <hip/hip_runtime.h>

typedef __attribute__((ext_vector_type(16))) _Float16 v16h;
typedef __attribute__((ext_vector_type(4)))  _Float16 v4h;
typedef __attribute__((ext_vector_type(8)))  float    v8f;
typedef __attribute__((ext_vector_type(4)))  int      v4i;

#define WMMA(a, b, c) \
  __builtin_amdgcn_wmma_f32_16x16x32_f16(false, (a), false, (b), (short)0, (c), false, false)

static constexpr int kB = 4, kS = 2048, kD = 512, kH = 8, kHD = 64;
static constexpr int kBS = kB * kS;  // 8192 tokens

// ---------------------------------------------------------------------------
// CDNA5 async global->LDS copy (16B per lane), tracked by ASYNCcnt.
// Builtin signature (from hipcc diagnostic): (AS1 int4* src, AS3 int4* dst, Ii, Ii)
// ---------------------------------------------------------------------------
__device__ __forceinline__ void async_cp16(void* lds_dst, const void* gsrc) {
#if __has_builtin(__builtin_amdgcn_global_load_async_to_lds_b128)
  __builtin_amdgcn_global_load_async_to_lds_b128(
      (__attribute__((address_space(1))) v4i*)gsrc,
      (__attribute__((address_space(3))) v4i*)lds_dst, 0, 0);
#else
  unsigned loff = (unsigned)(size_t)(__attribute__((address_space(3))) void*)lds_dst;
  asm volatile("global_load_async_to_lds_b128 %0, %1, off" ::"v"(loff), "v"(gsrc)
               : "memory");
#endif
}

#if __has_builtin(__builtin_amdgcn_s_wait_asynccnt)
#define WAIT_ASYNC(n) __builtin_amdgcn_s_wait_asynccnt(n)
#else
#define WAIT_ASYNC(n) asm volatile("s_wait_asynccnt %0" ::"i"(n) : "memory")
#endif

// A/B fragment load for v_wmma_f32_16x16x32_f16.
// A (16x32, MxK) from row-major [M][K]: lane&15 -> row, lane>>4 -> 16-wide K half.
// B (32x16, KxN) from [N][K] row-major (i.e. transposed B): identical addressing.
__device__ __forceinline__ const v16h* fragp(const _Float16* __restrict__ base, int ld,
                                             int row0, int col0, int lane) {
  return (const v16h*)(base + (size_t)(row0 + (lane & 15)) * ld + col0 + ((lane >> 4) << 4));
}

// ---------------------------------------------------------------------------
// f32 -> f16 conversion (vectorized x4)
// ---------------------------------------------------------------------------
__global__ __launch_bounds__(256) void cvt_f16_kernel(const float* __restrict__ src,
                                                      _Float16* __restrict__ dst, int n4) {
  int i = blockIdx.x * 256 + threadIdx.x;
  if (i < n4) {
    float4 f = ((const float4*)src)[i];
    v4h h = {(_Float16)f.x, (_Float16)f.y, (_Float16)f.z, (_Float16)f.w};
    ((v4h*)dst)[i] = h;
  }
}

// ---------------------------------------------------------------------------
// Y = X @ W^T + bias, X:[8192,512] f16, W:[512,512] f16 (row-major = [N][K]).
// 4 waves/block; each wave computes a 32x64 strip: 8 WMMAs per 6 frag loads.
// MODE 0: Q  -> [B,H,S,HD] f16, scaled by 1/sqrt(HD)
// MODE 1: K  -> [B,H,S,HD] f16
// MODE 2: V  -> [B,H,HD,S] f16 (transposed for P@V B-operand)
// MODE 3: O  -> [B,S,D] f32 (final output)
// ---------------------------------------------------------------------------
template <int MODE>
__global__ __launch_bounds__(128) void gemm_wmma_kernel(const _Float16* __restrict__ X,
                                                        const _Float16* __restrict__ W,
                                                        const float* __restrict__ bias,
                                                        void* __restrict__ Y) {
  const int lane = threadIdx.x & 31;
  const int wv   = threadIdx.x >> 5;
  const int m0   = blockIdx.x * 128 + wv * 32;
  const int n0   = blockIdx.y * 64;

  v8f acc[2][4] = {};
#pragma unroll 4
  for (int k0 = 0; k0 < kD; k0 += 32) {
    v16h a0 = *fragp(X, kD, m0, k0, lane);
    v16h a1 = *fragp(X, kD, m0 + 16, k0, lane);
#pragma unroll
    for (int j = 0; j < 4; ++j) {
      v16h b = *fragp(W, kD, n0 + 16 * j, k0, lane);
      acc[0][j] = WMMA(a0, b, acc[0][j]);
      acc[1][j] = WMMA(a1, b, acc[1][j]);
    }
  }

  // C layout: lanes 0-15 hold N=lane, M=vgpr; lanes 16-31 hold N=lane-16, M=8+vgpr
  const int rl = (lane >> 4) << 3;
  const int nl = lane & 15;
#pragma unroll
  for (int mi = 0; mi < 2; ++mi) {
#pragma unroll
    for (int j = 0; j < 4; ++j) {
#pragma unroll
      for (int v = 0; v < 8; ++v) {
        const int gr = m0 + 16 * mi + rl + v;  // token row
        const int gc = n0 + 16 * j + nl;       // output feature
        float val = acc[mi][j][v] + bias[gc];
        if (MODE == 0) val *= 0.125f;  // fold attention scale 1/sqrt(64) into Q
        if (MODE <= 1) {
          const int bb = gr >> 11, ss = gr & (kS - 1);
          const int hh = gc >> 6, dd = gc & (kHD - 1);
          ((_Float16*)Y)[((size_t)(bb * kH + hh) * kS + ss) * kHD + dd] = (_Float16)val;
        } else if (MODE == 2) {
          const int bb = gr >> 11, ss = gr & (kS - 1);
          const int hh = gc >> 6, dd = gc & (kHD - 1);
          ((_Float16*)Y)[((size_t)(bb * kH + hh) * kHD + dd) * kS + ss] = (_Float16)val;
        } else {
          ((float*)Y)[(size_t)gr * kD + gc] = val;
        }
      }
    }
  }
}

// ---------------------------------------------------------------------------
// Flash attention: grid (S/64 q-tiles, B*H). 4 waves/block, wave owns 16 q rows.
// K/V 64x64 tiles staged in LDS with double-buffered async copies (ASYNCcnt),
// online softmax; scale already folded into Qp. VpT is [B,H,HD,S].
// ---------------------------------------------------------------------------
__global__ __launch_bounds__(128) void flash_attn_kernel(const _Float16* __restrict__ Qp,
                                                         const _Float16* __restrict__ Kp,
                                                         const _Float16* __restrict__ VpT,
                                                         _Float16* __restrict__ ctx) {
  __shared__ _Float16 Ktile[2][64][64];  // [buf][key][hd]     8KB each
  __shared__ _Float16 Vtile[2][64][64];  // [buf][hd][key]     8KB each
  __shared__ _Float16 Plds[4][16][64];   // per-wave P bounce  8KB

  const int tid  = threadIdx.x;
  const int lane = tid & 31;
  const int wv   = tid >> 5;
  const int bh   = blockIdx.y;
  const int q0   = blockIdx.x * 64 + wv * 16;

  const _Float16* Qb = Qp + (size_t)bh * kS * kHD;
  const _Float16* Kb = Kp + (size_t)bh * kS * kHD;
  const _Float16* Vb = VpT + (size_t)bh * kHD * kS;

  // 8 x b128 async copies per thread per tile (K: 8KB, V: 8KB for 128 threads)
  auto issue_tile = [&](int kt, int buf) {
#pragma unroll
    for (int i = 0; i < 4; ++i) {
      const int idx = i * 128 + tid;   // 16B chunk id, 0..511
      const int row = idx >> 3;        // 0..63
      const int c8  = (idx & 7) << 3;  // half offset within row
      async_cp16(&Ktile[buf][row][c8], Kb + (size_t)(kt + row) * kHD + c8);
    }
#pragma unroll
    for (int i = 0; i < 4; ++i) {
      const int idx = i * 128 + tid;
      const int row = idx >> 3;        // hd
      const int c8  = (idx & 7) << 3;  // key offset
      async_cp16(&Vtile[buf][row][c8], Vb + (size_t)row * kS + kt + c8);
    }
  };

  const v16h aq0 = *fragp(Qb, kHD, q0, 0, lane);
  const v16h aq1 = *fragp(Qb, kHD, q0, 32, lane);

  v8f acc[4] = {};
  float mrow[8], lrow[8];
#pragma unroll
  for (int v = 0; v < 8; ++v) { mrow[v] = -1e30f; lrow[v] = 0.0f; }

  const int rl = (lane >> 4) << 3;
  const int nl = lane & 15;
  _Float16* Pw = &Plds[wv][0][0];

  issue_tile(0, 0);

  for (int ti = 0; ti < kS / 64; ++ti) {
    const int buf = ti & 1;
    if (ti + 1 < kS / 64) {
      issue_tile((ti + 1) * 64, buf ^ 1);
      WAIT_ASYNC(8);  // all but the 8 just-issued -> current tile resident
    } else {
      WAIT_ASYNC(0);
    }
    __syncthreads();

    const _Float16* Kt = &Ktile[buf][0][0];
    const _Float16* Vt = &Vtile[buf][0][0];

    // scores S = Q @ K^T (16 x 64) from LDS K tile
    v8f sc[4] = {};
#pragma unroll
    for (int j = 0; j < 4; ++j) {
      const _Float16* kp = Kt + (size_t)(16 * j + nl) * 64 + ((lane >> 4) << 4);
      v16h b0 = *(const v16h*)kp;
      sc[j] = WMMA(aq0, b0, sc[j]);
      v16h b1 = *(const v16h*)(kp + 32);
      sc[j] = WMMA(aq1, b1, sc[j]);
    }

    // row max across 64 keys: elementwise over tiles, then across 16 lanes
    float t[8];
#pragma unroll
    for (int v = 0; v < 8; ++v)
      t[v] = fmaxf(fmaxf(sc[0][v], sc[1][v]), fmaxf(sc[2][v], sc[3][v]));
#pragma unroll
    for (int off = 1; off < 16; off <<= 1)
#pragma unroll
      for (int v = 0; v < 8; ++v) t[v] = fmaxf(t[v], __shfl_xor(t[v], off, 32));

    float al[8], rs[8];
#pragma unroll
    for (int v = 0; v < 8; ++v) {
      const float mn = fmaxf(mrow[v], t[v]);
      al[v] = __expf(mrow[v] - mn);
      mrow[v] = mn;
      rs[v] = 0.0f;
    }

    // P = exp(S - m): accumulate row sums, write f16 P to LDS for A-layout reload
#pragma unroll
    for (int j = 0; j < 4; ++j)
#pragma unroll
      for (int v = 0; v < 8; ++v) {
        const float pv = __expf(sc[j][v] - mrow[v]);
        rs[v] += pv;
        Pw[(rl + v) * 64 + 16 * j + nl] = (_Float16)pv;
      }

#pragma unroll
    for (int off = 1; off < 16; off <<= 1)
#pragma unroll
      for (int v = 0; v < 8; ++v) rs[v] += __shfl_xor(rs[v], off, 32);

#pragma unroll
    for (int v = 0; v < 8; ++v) lrow[v] = lrow[v] * al[v] + rs[v];
#pragma unroll
    for (int j = 0; j < 4; ++j)
#pragma unroll
      for (int v = 0; v < 8; ++v) acc[j][v] *= al[v];

    __syncthreads();

    // reload P as A-fragments (16 x 64 over K=keys)
    const v16h ap0 = *(const v16h*)(Pw + nl * 64 + ((lane >> 4) << 4));
    const v16h ap1 = *(const v16h*)(Pw + nl * 64 + 32 + ((lane >> 4) << 4));

    // ctx += P @ V from LDS V tile ([hd][key] -> B operand contiguous along keys)
#pragma unroll
    for (int j = 0; j < 4; ++j) {
      const _Float16* vp = Vt + (size_t)(16 * j + nl) * 64 + ((lane >> 4) << 4);
      const v16h bv0 = *(const v16h*)vp;
      const v16h bv1 = *(const v16h*)(vp + 32);
      acc[j] = WMMA(ap0, bv0, acc[j]);
      acc[j] = WMMA(ap1, bv1, acc[j]);
    }
    __syncthreads();  // protect K/V buf and P region before next iteration
  }

  // normalize and store ctx as token-major [B,S,D] f16 for the output GEMM
  const int bb = bh >> 3, hh = bh & 7;
#pragma unroll
  for (int v = 0; v < 8; ++v) {
    const float inv = 1.0f / lrow[v];
    const int srow  = q0 + rl + v;
#pragma unroll
    for (int j = 0; j < 4; ++j)
      ctx[((size_t)bb * kS + srow) * kD + hh * kHD + 16 * j + nl] =
          (_Float16)(acc[j][v] * inv);
  }
}

// ---------------------------------------------------------------------------
extern "C" void kernel_launch(void* const* d_in, const int* in_sizes, int n_in,
                              void* d_out, int out_size, void* d_ws, size_t ws_size,
                              hipStream_t stream) {
  const float* q  = (const float*)d_in[0];
  const float* k  = (const float*)d_in[1];
  const float* v  = (const float*)d_in[2];
  const float* Wq = (const float*)d_in[3];
  const float* bq = (const float*)d_in[4];
  const float* Wk = (const float*)d_in[5];
  const float* bk = (const float*)d_in[6];
  const float* Wv = (const float*)d_in[7];
  const float* bv = (const float*)d_in[8];
  const float* Wo = (const float*)d_in[9];
  const float* bo = (const float*)d_in[10];

  char* p = (char*)d_ws;
  auto take = [&](size_t bytes) -> char* {
    char* r = p;
    p += (bytes + 255) & ~(size_t)255;
    return r;
  };
  const size_t tokElems = (size_t)kBS * kD;  // 8192*512
  const size_t wElems   = (size_t)kD * kD;

  _Float16* q16  = (_Float16*)take(tokElems * 2);
  _Float16* k16  = (_Float16*)take(tokElems * 2);
  _Float16* v16  = (_Float16*)take(tokElems * 2);
  _Float16* Wq16 = (_Float16*)take(wElems * 2);
  _Float16* Wk16 = (_Float16*)take(wElems * 2);
  _Float16* Wv16 = (_Float16*)take(wElems * 2);
  _Float16* Wo16 = (_Float16*)take(wElems * 2);
  _Float16* Qp   = (_Float16*)take(tokElems * 2);
  _Float16* Kp   = (_Float16*)take(tokElems * 2);
  _Float16* VpT  = (_Float16*)take(tokElems * 2);
  _Float16* ctx  = (_Float16*)take(tokElems * 2);

  auto cvt = [&](const float* src, _Float16* dst, size_t n) {
    int n4 = (int)(n / 4);
    cvt_f16_kernel<<<(n4 + 255) / 256, 256, 0, stream>>>(src, dst, n4);
  };
  cvt(q, q16, tokElems);
  cvt(k, k16, tokElems);
  cvt(v, v16, tokElems);
  cvt(Wq, Wq16, wElems);
  cvt(Wk, Wk16, wElems);
  cvt(Wv, Wv16, wElems);
  cvt(Wo, Wo16, wElems);

  dim3 ggrid(kBS / 128, kD / 64);
  gemm_wmma_kernel<0><<<ggrid, 128, 0, stream>>>(q16, Wq16, bq, (void*)Qp);
  gemm_wmma_kernel<1><<<ggrid, 128, 0, stream>>>(k16, Wk16, bk, (void*)Kp);
  gemm_wmma_kernel<2><<<ggrid, 128, 0, stream>>>(v16, Wv16, bv, (void*)VpT);

  flash_attn_kernel<<<dim3(kS / 64, kB * kH), 128, 0, stream>>>(Qp, Kp, VpT, ctx);

  gemm_wmma_kernel<3><<<ggrid, 128, 0, stream>>>(ctx, Wo16, bo, d_out);
}
